// Cross_ATT_85856396247440
// MI455X (gfx1250) — matrix operand order repaired and checked by reading, not compiled, not run on hardware
//
#include <hip/hip_runtime.h>

// MI455X / gfx1250, wave32, WMMA path.
// Pipeline: f32->bf16 convert, WMMA projection GEMMs (128x128 block tiles),
// two WMMA flash-attention passes (tmp = softmax(A k^T)v ; out = softmax(q A^T) tmp)
// with 64-wide KV tiles and async global->LDS staging for K.
// Workspace requirement: 64 MiB.

typedef __bf16 bf16;
typedef __attribute__((ext_vector_type(16))) __bf16 v16bf;
typedef __attribute__((ext_vector_type(8)))  __bf16 v8bf;
typedef __attribute__((ext_vector_type(4)))  __bf16 v4bf;
typedef __attribute__((ext_vector_type(8)))  float  v8f;
typedef __attribute__((ext_vector_type(4)))  float  v4f;

#define SCALE 0.125f   // HEAD_DIM^-0.5, D = 64

__device__ __forceinline__ v8f wmma_bf16(v16bf a, v16bf b, v8f c) {
    return __builtin_amdgcn_wmma_f32_16x16x32_bf16(
        /*neg_a=*/false, a, /*neg_b=*/false, b,
        /*c_mod=*/(short)0, c, /*reuse_a=*/false, /*reuse_b=*/false);
}

// A-fragment (16x32 bf16, row-major source with leading dim ld).
// lane L holds row (L&15); kbase = (L<16)?0:8.
// elems 0..7 -> K=kbase..kbase+7 ; elems 8..15 -> K=kbase+16..kbase+23.
__device__ __forceinline__ v16bf load_afrag(const bf16* src, int ld, int lane) {
    const int row = lane & 15;
    const int kb  = (lane < 16) ? 0 : 8;
    const bf16* rp = src + row * ld + kb;
    union { v16bf v; v8bf h[2]; } u;
    u.h[0] = *(const v8bf*)(rp);
    u.h[1] = *(const v8bf*)(rp + 16);
    return u.v;
}

// B-fragment (32x16 bf16) from an "N-major" source: src[n*ld + k].
// lane L holds col (L&15); kbase = (L<16)?0:8.
// elems 0..7 -> K=kbase..kbase+7 ; elems 8..15 -> K=kbase+16..kbase+23.
__device__ __forceinline__ v16bf load_bfrag(const bf16* src, int ld, int lane) {
    const int n  = lane & 15;
    const int kb = (lane < 16) ? 0 : 8;
    const bf16* rp = src + n * ld + kb;
    union { v16bf v; v8bf h[2]; } u;
    u.h[0] = *(const v8bf*)(rp);
    u.h[1] = *(const v8bf*)(rp + 16);
    return u.v;
}

// Async global -> LDS copy of 16 bytes (per-lane addresses), ASYNCcnt-tracked.
__device__ __forceinline__ void async_copy_b128(void* lds_dst, const void* gsrc) {
    unsigned lds_off = (unsigned)(unsigned long long)lds_dst;  // low 32 bits = LDS addr
    unsigned long long ga = (unsigned long long)gsrc;
    asm volatile("global_load_async_to_lds_b128 %0, %1, off"
                 :: "v"(lds_off), "v"(ga) : "memory");
}
__device__ __forceinline__ void wait_async() {
    asm volatile("s_wait_asynccnt 0" ::: "memory");
}

__global__ void cvt_f32_bf16(const float* __restrict__ in,
                             bf16* __restrict__ out, int n4) {
    int i = blockIdx.x * blockDim.x + threadIdx.x;
    if (i < n4) {
        v4f v = ((const v4f*)in)[i];
        v4bf r;
        r[0] = (bf16)v[0]; r[1] = (bf16)v[1];
        r[2] = (bf16)v[2]; r[3] = (bf16)v[3];
        ((v4bf*)out)[i] = r;
    }
}

// C[4096, Nout] = X[4096,1024] * W[Nout,1024]^T, bf16 output scattered into
// [B=2,H=16,N=2048,D=64] head layout; columns >= 1024 go to out1 (kv split).
// Block tile 128x128, 8 waves as 4x2, wave tile 32x64, K-step 32.
__global__ __launch_bounds__(256)
void gemm_proj(const bf16* __restrict__ X, const bf16* __restrict__ W,
               bf16* __restrict__ out0, bf16* __restrict__ out1) {
    __shared__ __align__(32) bf16 Xs[128 * 32];
    __shared__ __align__(32) bf16 Ws[128 * 32];

    const int tid  = threadIdx.x;
    const int lane = tid & 31;
    const int w    = tid >> 5;
    const int mw   = w >> 1;      // 0..3 (rows)
    const int nw   = w & 1;       // 0..1 (cols)
    const int m0   = blockIdx.x * 128;
    const int n0   = blockIdx.y * 128;

    v8f acc[2][4] = {};

    for (int k0 = 0; k0 < 1024; k0 += 32) {
        // Stage X and W tiles: 128x32 each = 512 v8bf chunks, 2 per thread.
#pragma unroll
        for (int i = 0; i < 2; ++i) {
            int cid = tid + 256 * i;
            int row = cid >> 2, col = (cid & 3) * 8;
            *(v8bf*)&Xs[row * 32 + col] =
                *(const v8bf*)&X[(size_t)(m0 + row) * 1024 + k0 + col];
            *(v8bf*)&Ws[row * 32 + col] =
                *(const v8bf*)&W[(size_t)(n0 + row) * 1024 + k0 + col];
        }
        __syncthreads();

        v16bf a0 = load_afrag(&Xs[(mw * 32 +  0) * 32], 32, lane);
        v16bf a1 = load_afrag(&Xs[(mw * 32 + 16) * 32], 32, lane);
#pragma unroll
        for (int j = 0; j < 4; ++j) {
            v16bf b = load_bfrag(&Ws[(nw * 64 + 16 * j) * 32], 32, lane);
            acc[0][j] = wmma_bf16(a0, b, acc[0][j]);
            acc[1][j] = wmma_bf16(a1, b, acc[1][j]);
        }
        __syncthreads();
    }

    // Epilogue: scatter into [B,H,N,D] bf16.
    const int rbase = (lane >= 16) ? 8 : 0;
#pragma unroll
    for (int i = 0; i < 2; ++i) {
#pragma unroll
        for (int j = 0; j < 4; ++j) {
            int c = n0 + nw * 64 + 16 * j + (lane & 15);
            bf16* op = out0;
            int cc = c;
            if (c >= 1024) { op = out1; cc = c - 1024; }
            int h = cc >> 6, d = cc & 63;
#pragma unroll
            for (int r = 0; r < 8; ++r) {
                int m = m0 + mw * 32 + 16 * i + rbase + r;
                int b = m >> 11, ntok = m & 2047;
                op[(((size_t)(b * 16 + h)) * 2048 + ntok) * 64 + d] =
                    (bf16)acc[i][j][r];
            }
        }
    }
}

// Flash attention: out = softmax(Q K^T * SCALE) @ V over [B*H=32, 2048, 64].
// Block = 128 query rows x one bh; 8 waves x 16 rows. KV tiles of 64:
// K async-staged row-major; V transposed into LDS; P round-trips per-wave LDS.
// outB != null -> bf16 [B,H,N,D]; outF != null -> f32 [B,N,C].
__global__ __launch_bounds__(256)
void flash_attn(const bf16* __restrict__ Q, const bf16* __restrict__ K,
                const bf16* __restrict__ V, bf16* __restrict__ outB,
                float* __restrict__ outF) {
    __shared__ __align__(32) bf16 Ks[64 * 64];      // [kv][d]
    __shared__ __align__(32) bf16 Vt[64 * 64];      // [d][kv] (transposed)
    __shared__ __align__(32) bf16 Pl[8 * 16 * 64];  // per-wave P scratch

    const int tid  = threadIdx.x;
    const int lane = tid & 31;
    const int w    = tid >> 5;
    const int bh   = blockIdx.y;
    const int q0   = blockIdx.x * 128 + w * 16;

    const bf16* Qb = Q + (size_t)bh * 2048 * 64;
    const bf16* Kb = K + (size_t)bh * 2048 * 64;
    const bf16* Vb = V + (size_t)bh * 2048 * 64;

    // Loop-invariant Q fragments: 16 rows x 64 d = two A-frags.
    v16bf aq0 = load_afrag(Qb + (size_t)q0 * 64 +  0, 64, lane);
    v16bf aq1 = load_afrag(Qb + (size_t)q0 * 64 + 32, 64, lane);

    v8f o[4] = {};
    float mrow[8], lrow[8];
#pragma unroll
    for (int r = 0; r < 8; ++r) { mrow[r] = -1e30f; lrow[r] = 0.0f; }

    bf16* Pw = &Pl[w * 16 * 64];
    const int rbase = (lane >= 16) ? 8 : 0;

    for (int kv0 = 0; kv0 < 2048; kv0 += 64) {
        // Stage 64x64 K (async -> LDS) and V (transposed): 2 chunks/thread each.
#pragma unroll
        for (int i = 0; i < 2; ++i) {
            int cid = tid + 256 * i;
            int kk = cid >> 3, dof = (cid & 7) * 8;
            async_copy_b128(&Ks[kk * 64 + dof],
                            &Kb[(size_t)(kv0 + kk) * 64 + dof]);
            v8bf vv = *(const v8bf*)&Vb[(size_t)(kv0 + kk) * 64 + dof];
#pragma unroll
            for (int e = 0; e < 8; ++e) Vt[(dof + e) * 64 + kk] = vv[e];
            if (kv0 + 64 < 2048) {  // prefetch next tile -> global_prefetch_b8
                __builtin_prefetch(&Kb[(size_t)(kv0 + 64 + kk) * 64 + dof], 0, 0);
                __builtin_prefetch(&Vb[(size_t)(kv0 + 64 + kk) * 64 + dof], 0, 0);
            }
        }
        wait_async();
        __syncthreads();

        // S = Q K^T : four 16x16 tiles across the 64 kv columns.
        v8f s[4];
#pragma unroll
        for (int j = 0; j < 4; ++j) {
            v8f z = {};
            z = wmma_bf16(aq0, load_bfrag(&Ks[16 * j * 64 +  0], 64, lane), z);
            z = wmma_bf16(aq1, load_bfrag(&Ks[16 * j * 64 + 32], 64, lane), z);
            s[j] = z;
        }

        // Online softmax; row reductions via shfl_xor in 16-lane halves.
#pragma unroll
        for (int r = 0; r < 8; ++r) {
            float v0 = s[0][r] * SCALE, v1 = s[1][r] * SCALE;
            float v2 = s[2][r] * SCALE, v3 = s[3][r] * SCALE;
            float mx = fmaxf(fmaxf(v0, v1), fmaxf(v2, v3));
            mx = fmaxf(mx, __shfl_xor(mx, 1, 32));
            mx = fmaxf(mx, __shfl_xor(mx, 2, 32));
            mx = fmaxf(mx, __shfl_xor(mx, 4, 32));
            mx = fmaxf(mx, __shfl_xor(mx, 8, 32));
            float mn   = fmaxf(mrow[r], mx);
            float corr = __expf(mrow[r] - mn);
            float p0 = __expf(v0 - mn), p1 = __expf(v1 - mn);
            float p2 = __expf(v2 - mn), p3 = __expf(v3 - mn);
            float rs = (p0 + p1) + (p2 + p3);
            rs += __shfl_xor(rs, 1, 32);
            rs += __shfl_xor(rs, 2, 32);
            rs += __shfl_xor(rs, 4, 32);
            rs += __shfl_xor(rs, 8, 32);
            lrow[r] = lrow[r] * corr + rs;
            mrow[r] = mn;
            o[0][r] = o[0][r] * corr; o[1][r] = o[1][r] * corr;
            o[2][r] = o[2][r] * corr; o[3][r] = o[3][r] * corr;
            // P (C-layout) -> LDS row-major 16x64, for A-frag reload.
            bf16* pp = &Pw[(rbase + r) * 64 + (lane & 15)];
            pp[0]  = (bf16)p0;
            pp[16] = (bf16)p1;
            pp[32] = (bf16)p2;
            pp[48] = (bf16)p3;
        }
        __syncthreads();

        // O += P @ V_tile. P: 16x64 -> two A-frags; V^T: 4x2 B-frags.
        {
            v16bf aP0 = load_afrag(&Pw[ 0], 64, lane);
            v16bf aP1 = load_afrag(&Pw[32], 64, lane);
#pragma unroll
            for (int t = 0; t < 4; ++t) {
                o[t] = wmma_bf16(aP0, load_bfrag(&Vt[16 * t * 64 +  0], 64, lane), o[t]);
                o[t] = wmma_bf16(aP1, load_bfrag(&Vt[16 * t * 64 + 32], 64, lane), o[t]);
            }
        }
        __syncthreads();
    }

    // Epilogue: normalize and store.
    const int b = bh >> 4, h = bh & 15;
#pragma unroll
    for (int t = 0; t < 4; ++t) {
        int col = 16 * t + (lane & 15);
#pragma unroll
        for (int r = 0; r < 8; ++r) {
            int row = q0 + rbase + r;
            float val = o[t][r] / lrow[r];
            if (outF) {
                outF[((size_t)(b * 2048 + row)) * 1024 + h * 64 + col] = val;
            } else {
                outB[((size_t)bh * 2048 + row) * 64 + col] = (bf16)val;
            }
        }
    }
}

extern "C" void kernel_launch(void* const* d_in, const int* in_sizes, int n_in,
                              void* d_out, int out_size, void* d_ws, size_t ws_size,
                              hipStream_t stream) {
    (void)in_sizes; (void)n_in; (void)out_size; (void)ws_size;

    const float* x   = (const float*)d_in[0];  // [2,2048,1024]
    const float* x2  = (const float*)d_in[1];  // [2,2048,1024]
    const float* Wq  = (const float*)d_in[2];  // [1024,1024]
    const float* Wa  = (const float*)d_in[3];  // [1024,1024]
    const float* Wkv = (const float*)d_in[4];  // [2048,1024]

    // Workspace layout (bf16), 64 MiB total.
    bf16* xb   = (bf16*)d_ws;
    bf16* x2b  = xb   + (size_t)4096 * 1024;
    bf16* Wqb  = x2b  + (size_t)4096 * 1024;
    bf16* Wab  = Wqb  + (size_t)1024 * 1024;
    bf16* Wkvb = Wab  + (size_t)1024 * 1024;
    bf16* qb   = Wkvb + (size_t)2048 * 1024;   // [B,H,N,D]
    bf16* Ab   = qb   + (size_t)4194304;
    bf16* kb   = Ab   + (size_t)4194304;
    bf16* vb   = kb   + (size_t)4194304;
    bf16* tb   = vb   + (size_t)4194304;       // tmp = softmax(A k^T) v

    cvt_f32_bf16<<<4096, 256, 0, stream>>>(x,   xb,   1048576);
    cvt_f32_bf16<<<4096, 256, 0, stream>>>(x2,  x2b,  1048576);
    cvt_f32_bf16<<<1024, 256, 0, stream>>>(Wq,  Wqb,  262144);
    cvt_f32_bf16<<<1024, 256, 0, stream>>>(Wa,  Wab,  262144);
    cvt_f32_bf16<<<2048, 256, 0, stream>>>(Wkv, Wkvb, 524288);

    gemm_proj<<<dim3(32,  8), 256, 0, stream>>>(xb,  Wqb,  qb, qb);
    gemm_proj<<<dim3(32,  8), 256, 0, stream>>>(x2b, Wab,  Ab, Ab);
    gemm_proj<<<dim3(32, 16), 256, 0, stream>>>(xb,  Wkvb, kb, vb);

    // Pass 1: tmp = softmax(A k^T * s) v   (bf16 out, head layout)
    flash_attn<<<dim3(16, 32), 256, 0, stream>>>(Ab, kb, vb, tb, nullptr);
    // Pass 2: out = softmax(q A^T * s) tmp (f32 out, [B,N,C])
    flash_attn<<<dim3(16, 32), 256, 0, stream>>>(qb, Ab, tb, nullptr,
                                                 (float*)d_out);
}
